// InductiveGCNwithIMGS_19980187861408
// MI455X (gfx1250) — matrix operand order repaired and checked by reading, not compiled, not run on hardware
//
#include <hip/hip_runtime.h>
#include <math.h>

typedef __attribute__((ext_vector_type(16))) _Float16 v16h;
typedef __attribute__((ext_vector_type(8)))  float    v8f;
typedef __attribute__((ext_vector_type(4)))  float    v4f;

__device__ __forceinline__ v8f wmma_f16(v16h a, v16h b, v8f c) {
  return __builtin_amdgcn_wmma_f32_16x16x32_f16(false, a, false, b, (short)0, c, false, false);
}

__device__ __forceinline__ v16h cvt_a_frag(const float* ap) {
  v4f a0 = *(const v4f*)(ap);
  v4f a1 = *(const v4f*)(ap + 4);
  v4f a2 = *(const v4f*)(ap + 16);
  v4f a3 = *(const v4f*)(ap + 20);
  v16h a;
#pragma unroll
  for (int e = 0; e < 4; ++e) {
    a[e]      = (_Float16)a0[e];
    a[4 + e]  = (_Float16)a1[e];
    a[8 + e]  = (_Float16)a2[e];
    a[12 + e] = (_Float16)a3[e];
  }
  return a;
}

// ---------------------------------------------------------------------------
// Pack B[K,N] (f32, leading dim ldb) into WMMA fragment order f16, zero-padded.
// ---------------------------------------------------------------------------
__global__ __launch_bounds__(32) void pack_b_k(
    const float* __restrict__ B, _Float16* __restrict__ Bp,
    int K, int N, int ldb)
{
  const int kstep = blockIdx.x, ntile = blockIdx.y, lane = threadIdx.x;
  const int n  = ntile * 16 + (lane & 15);
  const int kb = kstep * 32 + (lane >> 4) * 8;
  _Float16* out = Bp + (((size_t)kstep * gridDim.y + ntile) * 32 + lane) * 16;
#pragma unroll
  for (int e = 0; e < 16; ++e) {
    int kk = kb + ((e < 8) ? e : e + 8);
    float x = (n < N && kk < K) ? B[(size_t)kk * ldb + n] : 0.0f;
    out[e] = (_Float16)x;
  }
}

// ---------------------------------------------------------------------------
// WMMA GEMM: C = act(A[M,Kp] @ B + bias). Kp%32==0, ntiles = gridDim.y*4
// (exact). One wave -> 16x64 tile, no guards anywhere. ACT: 0/1(relu)/2(leaky)
// ---------------------------------------------------------------------------
template<int ACT>
__global__ __launch_bounds__(32) void gemm_wmma_t(
    const float* __restrict__ A, const _Float16* __restrict__ Bp,
    const float* __restrict__ bias, float* __restrict__ C,
    int Kp, int lda, int ldc)
{
  const int ntiles = (int)(gridDim.y << 2);
  const int row0 = blockIdx.x * 16;
  const int jt0  = blockIdx.y * 4;
  const int lane = threadIdx.x & 31;
  const int mA   = lane & 15;
  const int kgrp = lane >> 4;
  const int ksteps = Kp >> 5;
  const size_t bstride = (size_t)ntiles * 512;

  v8f acc0 = {}, acc1 = {}, acc2 = {}, acc3 = {};
  const float*    ap = A + (size_t)(row0 + mA) * lda + kgrp * 8;
  const _Float16* bb = Bp + ((size_t)jt0 * 32 + lane) * 16;

  for (int ks = 0; ks < ksteps; ++ks) {
    v16h a = cvt_a_frag(ap);
    __builtin_prefetch(bb + bstride, 0, 3);   // speculative, dropped on fault
    v16h b0 = *(const v16h*)(bb);
    v16h b1 = *(const v16h*)(bb + 512);
    v16h b2 = *(const v16h*)(bb + 1024);
    v16h b3 = *(const v16h*)(bb + 1536);
    acc0 = wmma_f16(a, b0, acc0);
    acc1 = wmma_f16(a, b1, acc1);
    acc2 = wmma_f16(a, b2, acc2);
    acc3 = wmma_f16(a, b3, acc3);
    ap += 32;
    bb += bstride;
  }

  const int mb = kgrp * 8;
  v8f* accs[4] = { &acc0, &acc1, &acc2, &acc3 };
#pragma unroll
  for (int j = 0; j < 4; ++j) {
    int col = (jt0 + j) * 16 + mA;
    float bv = bias ? bias[col] : 0.0f;
    v8f av = *accs[j];
#pragma unroll
    for (int r = 0; r < 8; ++r) {
      float x = av[r] + bv;
      if (ACT == 1) x = fmaxf(x, 0.0f);
      else if (ACT == 2) x = (x > 0.0f) ? x : 0.1f * x;
      C[(size_t)(row0 + mb + r) * ldc + col] = x;
    }
  }
}

// N==1 tail (decoder head): one n-tile, store column 0 only.
__global__ __launch_bounds__(32) void gemm_n1_k(
    const float* __restrict__ A, const _Float16* __restrict__ Bp,
    const float* __restrict__ bias, float* __restrict__ C,
    int Kp, int lda)
{
  const int row0 = blockIdx.x * 16;
  const int lane = threadIdx.x & 31;
  const int mA   = lane & 15;
  const int kgrp = lane >> 4;
  const int ksteps = Kp >> 5;
  v8f acc = {};
  const float*    ap = A + (size_t)(row0 + mA) * lda + kgrp * 8;
  const _Float16* bb = Bp + (size_t)lane * 16;
  for (int ks = 0; ks < ksteps; ++ks) {
    v16h a = cvt_a_frag(ap);
    v16h b = *(const v16h*)(bb);
    acc = wmma_f16(a, b, acc);
    ap += 32;
    bb += 512;
  }
  if (mA == 0) {
    float bv = bias[0];
    const int mb = kgrp * 8;
#pragma unroll
    for (int r = 0; r < 8; ++r) C[row0 + mb + r] = acc[r] + bv;
  }
}

// ---------------------------------------------------------------------------
// LayerNorm over last dim == 128
// ---------------------------------------------------------------------------
__global__ __launch_bounds__(128) void layernorm_k(
    const float* __restrict__ in, const float* __restrict__ g,
    const float* __restrict__ b, float* __restrict__ out)
{
  __shared__ float red[128];
  const int row = blockIdx.x, t = threadIdx.x;
  float v = in[(size_t)row * 128 + t];
  red[t] = v; __syncthreads();
  for (int s = 64; s > 0; s >>= 1) { if (t < s) red[t] += red[t + s]; __syncthreads(); }
  float mu = red[0] * (1.0f / 128.0f);
  __syncthreads();
  float d = v - mu;
  red[t] = d * d; __syncthreads();
  for (int s = 64; s > 0; s >>= 1) { if (t < s) red[t] += red[t + s]; __syncthreads(); }
  float var = red[0] * (1.0f / 128.0f);
  out[(size_t)row * 128 + t] = d * rsqrtf(var + 1e-5f) * g[t] + b[t];
}

// ---------------------------------------------------------------------------
// GCN helpers
// ---------------------------------------------------------------------------
__global__ void deg_init_k(float* deg, int n) {
  int i = blockIdx.x * 256 + threadIdx.x;
  if (i < n) deg[i] = 2.0f;
}
__global__ void deg_accum_k(const int* __restrict__ dst, float* deg, int e) {
  int i = blockIdx.x * 256 + threadIdx.x;
  if (i < e) atomicAdd(&deg[dst[i]], 1.0f);
}
__global__ void deg_rsqrt_k(const float* deg, float* dis, int n) {
  int i = blockIdx.x * 256 + threadIdx.x;
  if (i < n) dis[i] = rsqrtf(deg[i]);
}
__global__ __launch_bounds__(256) void self_add_k(
    float* __restrict__ agg, const float* __restrict__ h2,
    const float* __restrict__ dis, const float* __restrict__ gcn_b)
{
  int i = blockIdx.x * 256 + threadIdx.x;
  int n = i >> 8, c = i & 255;
  float di = dis[n];
  agg[i] += gcn_b[c] + 2.0f * di * di * h2[i];
}
__global__ __launch_bounds__(256) void edge_scatter_k(
    const int* __restrict__ src, const int* __restrict__ dst,
    const float* __restrict__ dis, const float* __restrict__ h2,
    float* __restrict__ agg)
{
  int e = blockIdx.x, c = threadIdx.x;
  int s = src[e], d = dst[e];
  float norm = dis[s] * dis[d];
  atomicAdd(&agg[(size_t)d * 256 + c], norm * h2[(size_t)s * 256 + c]);
}
__global__ __launch_bounds__(256) void elu_z_k(const float* __restrict__ agg,
                                               float* __restrict__ z)
{
  int i = blockIdx.x * 256 + threadIdx.x;
  int n = i >> 8, c = i & 255;
  float x = agg[i];
  z[(size_t)n * 384 + c] = (x > 0.0f) ? x : 0.1f * expm1f(fminf(x, 0.0f));
}

// ---------------------------------------------------------------------------
// CNN branch: one image per 256-thread block (8 waves). WMMA implicit GEMM,
// zero-padded LDS feature maps, division-free im2col (reg/LDS offset LUTs).
// Output [img, 64] with cols 36..63 zero (K-padded for fc1 GEMM).
// ---------------------------------------------------------------------------
__global__ __launch_bounds__(256) void cnn_k(
    const float* __restrict__ imgs,
    const float* __restrict__ w1, const float* __restrict__ b1,
    const float* __restrict__ w2, const float* __restrict__ b2,
    const float* __restrict__ w3, const float* __restrict__ b3,
    float* __restrict__ out64)
{
  __shared__ __align__(32) char smem[59648];
  _Float16* sImg   = (_Float16*)smem;             // 7500 halfs (phase1)
  _Float16* sC2    = (_Float16*)smem;             // 6912 halfs (phase2, reuse)
  _Float16* sP1    = (_Float16*)(smem + 15040);   // 16224 halfs (padded pool1)
  _Float16* sStrip = (_Float16*)(smem + 47488);   // 4608 halfs (phase1)
  _Float16* sB2p   = (_Float16*)(smem + 47488);   // 3584 halfs (phase2, reuse)
  float*    sP2    = (float*)   (smem + 47488);   // 1728 f32 (phase3, reuse)
  _Float16* sB1p   = (_Float16*)(smem + 56704);   // 1024 halfs
  float*    sT3    = (float*)   (smem + 56704);   // 144 f32 (phase3, reuse)
  int*      sLUT2  = (int*)     (smem + 58752);   // 224 ints: conv2 im2col offs

  const int img  = blockIdx.x;
  const int tid  = threadIdx.x;
  const int lane = tid & 31;
  const int wave = tid >> 5;
  const int mA   = lane & 15;
  const int kgrp = lane >> 4;
  const float* im = imgs + (size_t)img * 3 * 48 * 48;

  // zero padded buffers
  for (int t = tid; t < 7500; t += 256)  sImg[t] = (_Float16)0.0f;
  for (int t = tid; t < 16224; t += 256) sP1[t]  = (_Float16)0.0f;
  // conv1 weights pre-swizzled: [j(2)][lane(32)][e(16)]
  for (int t = tid; t < 1024; t += 256) {
    int j = t >> 9, ln = (t >> 4) & 31, e = t & 15;
    int n = j * 16 + (ln & 15);
    int kk = (ln >> 4) * 8 + ((e < 8) ? e : e + 8);
    sB1p[t] = (_Float16)((kk < 27 && n < 24) ? w1[n * 27 + kk] : 0.0f);
  }
  // conv2 im2col offset LUT: [g(2)][ks(7)][e(16)]
  for (int t = tid; t < 224; t += 256) {
    int g = t / 112, rem = t - g * 112;
    int ks = rem >> 4, e = rem & 15;
    int kk = ks * 32 + ((e < 8) ? 0 : 16) + g * 8 + (e & 7);
    int kkc = (kk < 216) ? kk : 0;            // B panel zero past 216
    int ci = kkc / 9, rr = kkc - ci * 9;
    int r = rr / 3, s = rr - r * 3;
    sLUT2[t] = ci * 676 + r * 26 + s;
  }
  __syncthreads();
  // image interior -> padded LDS [3][50][50]
  for (int t = tid; t < 3 * 48 * 48; t += 256) {
    int ci = t / 2304, rem = t - ci * 2304;
    int iy = rem / 48, ix = rem - iy * 48;
    sImg[ci * 2500 + (iy + 1) * 50 + (ix + 1)] = (_Float16)im[t];
  }

  // conv1 im2col base offsets in registers (once per wave)
  int base1[16];
#pragma unroll
  for (int e = 0; e < 16; ++e) {
    int kk = ((e < 8) ? 0 : 16) + kgrp * 8 + (e & 7);
    int kkc = (kk < 27) ? kk : 0;             // B panel zero past 27
    int ci = kkc / 9, rr = kkc - ci * 9;
    int r = rr / 3, s = rr - r * 3;
    base1[e] = ci * 2500 + r * 50 + s;
  }
  __syncthreads();

  // ---- conv1 (+relu+pool) in 12 strips of 4 output rows ----
  for (int strip = 0; strip < 12; ++strip) {
    const int y0 = strip * 4;
    for (int tile = wave; tile < 12; tile += 8) {   // 4*48 px -> 12 tiles
      int p  = tile * 16 + mA;
      int py = y0 + p / 48;
      int px = p % 48;
      int pix = py * 50 + px;
      v16h a;
#pragma unroll
      for (int e = 0; e < 16; ++e) a[e] = sImg[base1[e] + pix];
      const int mb = kgrp * 8;
#pragma unroll
      for (int j = 0; j < 2; ++j) {
        v16h b = *(const v16h*)(sB1p + (j * 32 + lane) * 16);
        v8f acc = {};
        acc = wmma_f16(a, b, acc);
        int n = j * 16 + mA;
        if (n < 24) {
          float bias = b1[n];
#pragma unroll
          for (int r = 0; r < 8; ++r)
            sStrip[n * 192 + tile * 16 + mb + r] = (_Float16)fmaxf(acc[r] + bias, 0.0f);
        }
      }
    }
    __syncthreads();
    for (int idx = tid; idx < 24 * 2 * 24; idx += 256) {
      int ch = idx / 48, rem = idx - ch * 48;
      int ry = rem / 24, cx = rem - ry * 24;
      int base = ch * 192 + ry * 96 + cx * 2;
      float m = fmaxf(fmaxf((float)sStrip[base], (float)sStrip[base + 1]),
                      fmaxf((float)sStrip[base + 48], (float)sStrip[base + 49]));
      sP1[ch * 676 + (strip * 2 + ry + 1) * 26 + (cx + 1)] = (_Float16)m;
    }
    __syncthreads();
  }

  // conv2 weights pre-swizzled into R2 (sStrip dead): [ks(7)][lane(32)][e(16)]
  for (int t = tid; t < 3584; t += 256) {
    int ks = t >> 9, ln = (t >> 4) & 31, e = t & 15;
    int n = ln & 15;
    int kk = ks * 32 + (ln >> 4) * 8 + ((e < 8) ? e : e + 8);
    sB2p[t] = (_Float16)((kk < 216 && n < 12) ? w2[n * 216 + kk] : 0.0f);
  }
  __syncthreads();

  // ---- conv2 (+relu): 24ch 24x24 -> 12ch 24x24, out to sC2 ----
  const int* lut2 = sLUT2 + kgrp * 112;
  for (int tile = wave; tile < 36; tile += 8) {     // 576 px -> 36 tiles
    int p  = tile * 16 + mA;
    int py = p / 24, px = p - py * 24;
    int pix = py * 26 + px;
    v8f acc = {};
#pragma unroll
    for (int ks = 0; ks < 7; ++ks) {
      v16h a;
#pragma unroll
      for (int e = 0; e < 16; ++e) a[e] = sP1[lut2[ks * 16 + e] + pix];
      v16h b = *(const v16h*)(sB2p + (ks * 32 + lane) * 16);
      acc = wmma_f16(a, b, acc);
    }
    const int mb = kgrp * 8;
    if (mA < 12) {
      float bias = b2[mA];
#pragma unroll
      for (int r = 0; r < 8; ++r)
        sC2[mA * 576 + tile * 16 + mb + r] = (_Float16)fmaxf(acc[r] + bias, 0.0f);
    }
  }
  __syncthreads();
  // pool conv2 -> sP2
  for (int idx = tid; idx < 12 * 12 * 12; idx += 256) {
    int ch = idx / 144, rem = idx - ch * 144;
    int ry = rem / 12, cx = rem - ry * 12;
    int base = ch * 576 + ry * 48 + cx * 2;
    sP2[idx] = fmaxf(fmaxf((float)sC2[base], (float)sC2[base + 1]),
                     fmaxf((float)sC2[base + 24], (float)sC2[base + 25]));
  }
  __syncthreads();
  // conv3 (1 out ch) + relu, VALU
  if (tid < 144) {
    int py = tid / 12, px = tid - py * 12;
    float sum = b3[0];
    for (int ci = 0; ci < 12; ++ci)
      for (int r = 0; r < 3; ++r) {
        int iy = py + r - 1;
        if (iy < 0 || iy >= 12) continue;
        for (int s = 0; s < 3; ++s) {
          int ix = px + s - 1;
          if (ix < 0 || ix >= 12) continue;
          sum += w3[ci * 9 + r * 3 + s] * sP2[ci * 144 + iy * 12 + ix];
        }
      }
    sT3[tid] = fmaxf(sum, 0.0f);
  }
  __syncthreads();
  if (tid < 64) {
    float v = 0.0f;
    if (tid < 36) {
      int ry = tid / 6, cx = tid - ry * 6;
      int base = ry * 24 + cx * 2;
      v = fmaxf(fmaxf(sT3[base], sT3[base + 1]), fmaxf(sT3[base + 12], sT3[base + 13]));
    }
    out64[(size_t)img * 64 + tid] = v;
  }
}

// ---------------------------------------------------------------------------
extern "C" void kernel_launch(void* const* d_in, const int* in_sizes, int n_in,
                              void* d_out, int out_size, void* d_ws, size_t ws_size,
                              hipStream_t stream)
{
  const float* x      = (const float*)d_in[0];
  const float* imgs   = (const float*)d_in[1];
  const int*   edges  = (const int*)d_in[2];
  const float* enc_w1 = (const float*)d_in[3];
  const float* enc_b1 = (const float*)d_in[4];
  const float* enc_w2 = (const float*)d_in[5];
  const float* enc_b2 = (const float*)d_in[6];
  const float* ln_g   = (const float*)d_in[7];
  const float* ln_b   = (const float*)d_in[8];
  const float* gcn_w  = (const float*)d_in[9];
  const float* gcn_b  = (const float*)d_in[10];
  const float* skip_w = (const float*)d_in[11];
  const float* skip_b = (const float*)d_in[12];
  const float* c1w = (const float*)d_in[13];
  const float* c1b = (const float*)d_in[14];
  const float* c2w = (const float*)d_in[15];
  const float* c2b = (const float*)d_in[16];
  const float* c3w = (const float*)d_in[17];
  const float* c3b = (const float*)d_in[18];
  const float* fc1_w = (const float*)d_in[19];
  const float* fc1_b = (const float*)d_in[20];
  const float* fc2_w = (const float*)d_in[21];
  const float* fc2_b = (const float*)d_in[22];
  const float* dec_w1 = (const float*)d_in[23];
  const float* dec_b1 = (const float*)d_in[24];
  const float* dec_w2 = (const float*)d_in[25];
  const float* dec_b2 = (const float*)d_in[26];

  const int Nn = in_sizes[0] / 128;   // 4096 nodes
  const int E  = in_sizes[2] / 2;     // 262144 edges
  const int Mt = Nn / 16;

  const int* srcp = edges;
  const int* dstp = edges + E;

  float* ws  = (float*)d_ws;
  float* t1  = ws;                      // [Nn,128]
  float* t2  = t1  + (size_t)Nn * 128;
  float* h   = t2  + (size_t)Nn * 128;
  float* h2  = h   + (size_t)Nn * 128;  // [Nn,256]
  float* agg = h2  + (size_t)Nn * 256;  // [Nn,256]
  float* deg = agg + (size_t)Nn * 256;
  float* dis = deg + (size_t)Nn;
  float* z   = dis + (size_t)Nn;        // [Nn,384]
  float* p3  = z   + (size_t)Nn * 384;  // [Nn,64]
  float* fcb = p3  + (size_t)Nn * 64;   // [Nn,256]
  float* db  = fcb + (size_t)Nn * 256;  // [Nn,256]

  _Float16* bpool = (_Float16*)(db + (size_t)Nn * 256);
  size_t boff = 0;
  auto packed = [&](int Kp, int ntiles) {
    _Float16* p = bpool + boff;
    boff += (size_t)(Kp >> 5) * ntiles * 32 * 16;
    return p;
  };
  _Float16* Bp_enc1 = packed(128, 8);
  _Float16* Bp_enc2 = packed(128, 8);
  _Float16* Bp_gcn  = packed(128, 16);
  _Float16* Bp_skip = packed(128, 16);
  _Float16* Bp_fc1  = packed(64, 16);
  _Float16* Bp_fc2  = packed(256, 8);
  _Float16* Bp_dec1 = packed(384, 16);
  _Float16* Bp_dec2 = packed(256, 1);

  dim3 w32(32);
  pack_b_k<<<dim3(4, 8),  w32, 0, stream>>>(enc_w1, Bp_enc1, 128, 128, 128);
  pack_b_k<<<dim3(4, 8),  w32, 0, stream>>>(enc_w2, Bp_enc2, 128, 128, 128);
  pack_b_k<<<dim3(4, 16), w32, 0, stream>>>(gcn_w,  Bp_gcn,  128, 256, 256);
  pack_b_k<<<dim3(4, 16), w32, 0, stream>>>(skip_w, Bp_skip, 128, 256, 256);
  pack_b_k<<<dim3(2, 16), w32, 0, stream>>>(fc1_w,  Bp_fc1,  36,  256, 256);
  pack_b_k<<<dim3(8, 8),  w32, 0, stream>>>(fc2_w,  Bp_fc2,  256, 128, 128);
  pack_b_k<<<dim3(12, 16),w32, 0, stream>>>(dec_w1, Bp_dec1, 384, 256, 256);
  pack_b_k<<<dim3(8, 1),  w32, 0, stream>>>(dec_w2, Bp_dec2, 256, 1,   1);

  // encoder
  gemm_wmma_t<1><<<dim3(Mt, 2), w32, 0, stream>>>(x,  Bp_enc1, enc_b1, t1, 128, 128, 128);
  gemm_wmma_t<0><<<dim3(Mt, 2), w32, 0, stream>>>(t1, Bp_enc2, enc_b2, t2, 128, 128, 128);
  layernorm_k<<<Nn, 128, 0, stream>>>(t2, ln_g, ln_b, h);

  // GCN block
  gemm_wmma_t<0><<<dim3(Mt, 4), w32, 0, stream>>>(h, Bp_gcn,  nullptr, h2,  128, 128, 256);
  gemm_wmma_t<0><<<dim3(Mt, 4), w32, 0, stream>>>(h, Bp_skip, skip_b,  agg, 128, 128, 256);
  deg_init_k<<<(Nn + 255) / 256, 256, 0, stream>>>(deg, Nn);
  deg_accum_k<<<(E + 255) / 256, 256, 0, stream>>>(dstp, deg, E);
  deg_rsqrt_k<<<(Nn + 255) / 256, 256, 0, stream>>>(deg, dis, Nn);
  self_add_k<<<Nn, 256, 0, stream>>>(agg, h2, dis, gcn_b);
  edge_scatter_k<<<E, 256, 0, stream>>>(srcp, dstp, dis, h2, agg);
  elu_z_k<<<Nn, 256, 0, stream>>>(agg, z);

  // CNN branch
  cnn_k<<<Nn, 256, 0, stream>>>(imgs, c1w, c1b, c2w, c2b, c3w, c3b, p3);
  gemm_wmma_t<1><<<dim3(Mt, 4), w32, 0, stream>>>(p3,  Bp_fc1, fc1_b, fcb,     64,  64,  256);
  gemm_wmma_t<0><<<dim3(Mt, 2), w32, 0, stream>>>(fcb, Bp_fc2, fc2_b, z + 256, 256, 256, 384);

  // decoder
  gemm_wmma_t<2><<<dim3(Mt, 4), w32, 0, stream>>>(z,  Bp_dec1, dec_b1, db, 384, 384, 256);
  gemm_n1_k<<<dim3(Mt), w32, 0, stream>>>(db, Bp_dec2, dec_b2, (float*)d_out, 256, 256);
}
